// MultiHeadAttention_89575837925997
// MI455X (gfx1250) — matrix operand order repaired and checked by reading, not compiled, not run on hardware
//
#include <hip/hip_runtime.h>
#include <hip/hip_bf16.h>

typedef __attribute__((ext_vector_type(16))) _Float16 v16h;
typedef __attribute__((ext_vector_type(8)))  float    v8f;
typedef __attribute__((ext_vector_type(4)))  int      v4i;
typedef __attribute__((ext_vector_type(4)))  float    v4f;

#define N_TOK 8192
#define NF    64
#define NH    4
#define DV    16

// 1/sqrt(64) * log2(e): fold both the attention scale and the exp->exp2
// conversion into Q at projection time (softmax is invariant to the shift,
// and exp2(s*log2e - m*log2e) == exp(s - m)).
#define QSCALE 0.18033688011112042f

union Pack16 {
    v16h v;
    _Float16 h[16];
    int  i[8];
    v4i  q[2];
};

__device__ __forceinline__ v4i ld16B(const _Float16* p) {
    return *reinterpret_cast<const v4i*>(p);
}

// Build a v16h A/B operand from two contiguous 16-byte chunks.
__device__ __forceinline__ v16h ldA(const _Float16* lo, const _Float16* hi) {
    Pack16 u;
    u.q[0] = ld16B(lo);
    u.q[1] = ld16B(hi);
    return u.v;
}

// Raw v_exp_f32 (no denormal-guard expansion): args are always <= 0 and the
// sub-2^-126 tail of a softmax is semantically zero.
__device__ __forceinline__ float fast_exp2(float x) {
    return __builtin_amdgcn_exp2f(x);
}

// ---------------------------------------------------------------------------
// Phase 1: per-head projections  Q = QSCALE*(x@Wq+bq), K = x@Wk+bk, V = x@Wv+bv
// Q,K stored [H][N][64] f16 row-major; V stored transposed Vt[H][16][N] f16.
// One thread per (h, n, e); ~0.6 GFLOP total, negligible vs attention.
// ---------------------------------------------------------------------------
__global__ __launch_bounds__(256) void mha_proj(
    const float* __restrict__ x,
    const float* __restrict__ Wq, const float* __restrict__ bq,
    const float* __restrict__ Wk, const float* __restrict__ bk,
    const float* __restrict__ Wv, const float* __restrict__ bv,
    _Float16* __restrict__ Qh, _Float16* __restrict__ Kh,
    _Float16* __restrict__ Vt)
{
    int tid = blockIdx.x * blockDim.x + threadIdx.x;   // H*N*64 = 2^21 threads
    int e = tid & (NF - 1);
    int n = (tid >> 6) & (N_TOK - 1);
    int h = tid >> 19;

    const float* xr = x  + (size_t)n * NF;
    const float* wq = Wq + (size_t)h * NF * NF + e;     // stride NF over f
    const float* wk = Wk + (size_t)h * NF * NF + e;
    const float* wv = Wv + (size_t)h * NF * DV + (e & (DV - 1));
    bool dov = (e < DV);

    float accq = bq[h * NF + e];
    float acck = bk[h * NF + e];
    float accv = 0.f;
    #pragma unroll
    for (int f = 0; f < NF; ++f) {
        float xv = xr[f];
        accq = fmaf(xv, wq[f * NF], accq);
        acck = fmaf(xv, wk[f * NF], acck);
        if (dov) accv = fmaf(xv, wv[f * DV], accv);
    }
    size_t hn = (size_t)h * N_TOK + n;
    Qh[hn * NF + e] = (_Float16)(accq * QSCALE);
    Kh[hn * NF + e] = (_Float16)acck;
    if (dov) {
        accv += bv[h * DV + e];
        Vt[((size_t)h * DV + e) * N_TOK + n] = (_Float16)accv;
    }
}

// ---------------------------------------------------------------------------
// Phase 2: flash attention. One wave = one (head, 16-query tile), streaming
// keys in blocks of 32.  S^T = K_tile x Q^T via v_wmma_f32_16x16x32_f16,
// online softmax (base-2) in registers, O^T += V^T x P^T via one more WMMA.
// The 8 waves of a block share one head + 8 consecutive query tiles, so they
// stream the SAME K/V blocks -> WGP$ serves ~7/8 of the loads.
// ---------------------------------------------------------------------------
__global__ __launch_bounds__(256) void mha_attn(
    const _Float16* __restrict__ Qh,
    const _Float16* __restrict__ Kh,
    const _Float16* __restrict__ Vt,
    float* __restrict__ out)
{
    const int lane = threadIdx.x & 31;
    const int wid  = blockIdx.x * 8 + (threadIdx.x >> 5);
    const int qt   = wid & 511;           // consecutive within a block
    const int h    = wid >> 9;            // same head for all 8 waves of block
    const int q    = lane & 15;
    const int hi   = lane >> 4;           // half-wave id
    const int n    = qt * 16 + q;

    // B operand: Q^T (dims x queries). Lane = query column; lane-half selects
    // dim range. Two 8-half runs per lane per 32-dim chunk (dims 8*hi.., 16+8*hi..).
    const _Float16* qrow = Qh + ((size_t)h * N_TOK + n) * NF;
    const v16h qb0 = ldA(qrow +      hi * 8, qrow +      hi * 8 + 16);
    const v16h qb1 = ldA(qrow + 32 + hi * 8, qrow + 32 + hi * 8 + 16);

    // A operand bases: K rows (lane = key row within tile) and V^T rows
    // (lane = output dim).
    const _Float16* kbase = Kh + ((size_t)h * N_TOK + q) * NF + 8 * hi;
    const _Float16* vbase = Vt + ((size_t)h * DV + q) * N_TOK + 16 * hi;

    float m = -INFINITY, lsum = 0.f;
    v8f o = {};

    for (int kb = 0; kb < N_TOK; kb += 32) {
        const _Float16* kr  = kbase + (size_t)kb * NF;   // keys kb..kb+15
        const _Float16* kr2 = kr + 16 * NF;              // keys kb+16..kb+31

        // Prefetch the K/V blocks two iterations ahead. Locality 3 -> nearest
        // scope, so the lines land in WGP$ where all 8 waves share them.
        __builtin_prefetch(kr  + 64 * NF, 0, 3);
        __builtin_prefetch(kr2 + 64 * NF, 0, 3);
        __builtin_prefetch(vbase + kb + 64, 0, 3);

        v8f zero = {};
        // S^T tile0: rows = keys kb..kb+15, cols = queries (K-dim 64 = 2x32)
        v8f c0 = __builtin_amdgcn_wmma_f32_16x16x32_f16(
            false, ldA(kr, kr + 16), false, qb0, (short)0, zero, false, false);
        c0 = __builtin_amdgcn_wmma_f32_16x16x32_f16(
            false, ldA(kr + 32, kr + 48), false, qb1, (short)0, c0, false, false);
        // S^T tile1: rows = keys kb+16..kb+31
        v8f c1 = __builtin_amdgcn_wmma_f32_16x16x32_f16(
            false, ldA(kr2, kr2 + 16), false, qb0, (short)0, zero, false, false);
        c1 = __builtin_amdgcn_wmma_f32_16x16x32_f16(
            false, ldA(kr2 + 32, kr2 + 48), false, qb1, (short)0, c1, false, false);

        // Online softmax in log2 units. This lane holds 16 of the 32 key
        // scores for query q; the other 16 live in the partner lane (xor 16).
        float mloc = c0[0];
        #pragma unroll
        for (int r = 1; r < 8; ++r) mloc = fmaxf(mloc, c0[r]);
        #pragma unroll
        for (int r = 0; r < 8; ++r) mloc = fmaxf(mloc, c1[r]);
        mloc = fmaxf(mloc, __shfl_xor(mloc, 16, 32));
        float mnew  = fmaxf(m, mloc);
        float alpha = fast_exp2(m - mnew);       // 0 on first block (m=-inf)

        Pack16 pb;
        float psum = 0.f;
        #pragma unroll
        for (int r = 0; r < 8; ++r) {
            float p = fast_exp2(c0[r] - mnew);
            psum += p;
            pb.h[r] = (_Float16)p;
        }
        #pragma unroll
        for (int r = 0; r < 8; ++r) {
            float p = fast_exp2(c1[r] - mnew);
            psum += p;
            pb.h[8 + r] = (_Float16)p;
        }
        psum += __shfl_xor(psum, 16, 32);
        lsum = lsum * alpha + psum;
        m = mnew;

        // Build B operand P^T (keys x queries) from the S^T C-layout:
        // exchange packed f16 pairs with partner lane, select per half-wave.
        int pr[8];
        #pragma unroll
        for (int j = 0; j < 8; ++j) pr[j] = __shfl_xor(pb.i[j], 16, 32);
        Pack16 bb;
        #pragma unroll
        for (int j = 0; j < 4; ++j) {
            bb.i[j]     = hi ? pr[4 + j]  : pb.i[j];
            bb.i[4 + j] = hi ? pb.i[4 + j] : pr[j];
        }

        // O^T = alpha*O^T + V^T x P^T  (alpha is per-query = per-lane)
        v8f oc;
        #pragma unroll
        for (int r = 0; r < 8; ++r) oc[r] = o[r] * alpha;
        const _Float16* vr = vbase + kb;
        o = __builtin_amdgcn_wmma_f32_16x16x32_f16(
            false, ldA(vr, vr + 8), false, bb.v, (short)0, oc, false, false);
    }

    // Normalize and write head slice: out[n][h*16 + d], 8 contiguous floats.
    float inv = 1.0f / lsum;
    float* op = out + (size_t)n * NF + h * DV + 8 * hi;
    v4f r0, r1;
    #pragma unroll
    for (int r = 0; r < 4; ++r) { r0[r] = o[r] * inv; r1[r] = o[4 + r] * inv; }
    *reinterpret_cast<v4f*>(op)     = r0;
    *reinterpret_cast<v4f*>(op + 4) = r1;
}

extern "C" void kernel_launch(void* const* d_in, const int* in_sizes, int n_in,
                              void* d_out, int out_size, void* d_ws, size_t ws_size,
                              hipStream_t stream) {
    const float* x  = (const float*)d_in[0];
    const float* Wq = (const float*)d_in[1];
    const float* bq = (const float*)d_in[2];
    const float* Wk = (const float*)d_in[3];
    const float* bk = (const float*)d_in[4];
    const float* Wv = (const float*)d_in[5];
    const float* bv = (const float*)d_in[6];
    float* out = (float*)d_out;

    // Workspace: Qh (4 MiB) | Kh (4 MiB) | Vt (1 MiB), all f16.
    char* ws = (char*)d_ws;
    _Float16* Qh = (_Float16*)ws;
    _Float16* Kh = (_Float16*)(ws + (size_t)NH * N_TOK * NF * sizeof(_Float16));
    _Float16* Vt = (_Float16*)(ws + (size_t)2 * NH * N_TOK * NF * sizeof(_Float16));

    mha_proj<<<(NH * N_TOK * NF) / 256, 256, 0, stream>>>(
        x, Wq, bq, Wk, bk, Wv, bv, Qh, Kh, Vt);

    // 2048 waves total: 8 waves/block -> 256 blocks.
    mha_attn<<<(NH * (N_TOK / 16)) / 8, 256, 0, stream>>>(Qh, Kh, Vt, out);
}
